// Model_87479893885134
// MI455X (gfx1250) — matrix operand order repaired and checked by reading, not compiled, not run on hardware
//
#include <hip/hip_runtime.h>
#include <hip/hip_bf16.h>

typedef __attribute__((ext_vector_type(16))) _Float16     v16h;
typedef __attribute__((ext_vector_type(8)))  float        v8f;
typedef __attribute__((ext_vector_type(4)))  unsigned int u32x4;
typedef __attribute__((ext_vector_type(8)))  int          i32x8;
typedef __attribute__((ext_vector_type(4)))  int          i32x4;

#define BDIM    32
#define LLEN    512
#define NCIN    32
#define DMODEL  128
#define NSTATE  64
#define ELAYERS 3
#define PLEN    16
#define STRIDE  8
#define NPRED   96
#define PNUM    64
#define BC      (BDIM*NCIN)   /* 1024 */
#define MROWS   (BC*PNUM)     /* 65536 */

// ---------------- RevIN statistics: mean/std over L per (b,c) ----------------
__global__ __launch_bounds__(256) void stats_kernel(const float* __restrict__ x,
                                                    float* __restrict__ means,
                                                    float* __restrict__ stds) {
    int bc = blockIdx.x;               // 0..1023
    int b = bc >> 5, c = bc & 31;
    const float* base = x + (size_t)b * LLEN * NCIN + c;
    float s = 0.f, ss = 0.f;
    for (int t = threadIdx.x; t < LLEN; t += 256) {
        float v = base[(size_t)t * NCIN];
        s += v; ss += v * v;
    }
    __shared__ float sh1[256], sh2[256];
    sh1[threadIdx.x] = s; sh2[threadIdx.x] = ss;
    __syncthreads();
    for (int o = 128; o; o >>= 1) {
        if (threadIdx.x < o) {
            sh1[threadIdx.x] += sh1[threadIdx.x + o];
            sh2[threadIdx.x] += sh2[threadIdx.x + o];
        }
        __syncthreads();
    }
    if (threadIdx.x == 0) {
        float m = sh1[0] * (1.f / LLEN);
        float var = sh2[0] * (1.f / LLEN) - m * m;
        means[bc] = m;
        stds[bc] = sqrtf(var + 1e-5f);
    }
}

// ---------------- fp32 -> fp16 weight conversion ----------------
__global__ void cvt_kernel(const float* __restrict__ src, _Float16* __restrict__ dst, int n) {
    int i = blockIdx.x * blockDim.x + threadIdx.x;
    if (i < n) dst[i] = (_Float16)src[i];
}

// ---------------- positional embedding table (P x D) ----------------
__global__ void pe_kernel(float* __restrict__ pe) {
    int i = blockIdx.x * blockDim.x + threadIdx.x;
    if (i >= PNUM * DMODEL) return;
    int p = i >> 7, d = i & 127;
    float div = expf(-(float)(d & ~1) * (logf(10000.f) / (float)DMODEL));
    pe[i] = (d & 1) ? cosf((float)p * div) : sinf((float)p * div);
}

// ---------------- S4D kernel precompute: K[e][h][p] ----------------
__global__ void s4_kernel_precompute(const float* __restrict__ log_dt,
                                     const float* __restrict__ arel,
                                     const float* __restrict__ aim,
                                     const float* __restrict__ cre,
                                     const float* __restrict__ cim,
                                     float* __restrict__ Kbuf) {
    int i = blockIdx.x * blockDim.x + threadIdx.x;
    if (i >= ELAYERS * DMODEL * PNUM) return;
    int p = i & 63;
    int h = (i >> 6) & 127;
    int e = i >> 13;
    float dt = expf(log_dt[e * DMODEL + h]);
    int off = (e * DMODEL + h) * NSTATE;
    const float* ar = arel + off;
    const float* ai = aim + off;
    const float* cr = cre + off;
    const float* ci = cim + off;
    float acc = 0.f;
    for (int n = 0; n < NSTATE; n++) {
        float are = -expf(ar[n]);
        float aiv = ai[n];
        float zr = are * dt, zi = aiv * dt;           // dtA
        float er = expf(zr);
        float nr = er * cosf(zi) - 1.f;               // exp(dtA)-1
        float ni = er * sinf(zi);
        float den = are * are + aiv * aiv;
        float tr = (nr * are + ni * aiv) / den;       // (exp(dtA)-1)/A
        float ti = (ni * are - nr * aiv) / den;
        float crv = cr[n], civ = ci[n];
        float c2r = crv * tr - civ * ti;              // coef = Ct * t
        float c2i = crv * ti + civ * tr;
        float ep = expf(zr * (float)p);               // exp(dtA * p)
        float wr = ep * cosf(zi * (float)p);
        float wi = ep * sinf(zi * (float)p);
        acc += c2r * wr - c2i * wi;
    }
    Kbuf[i] = 2.f * acc;                              // layout [e][h][p]
}

// ---------------- patch embedding GEMM: (65536 x 16pad32) @ (16 x 128) + pos ----------------
__global__ __launch_bounds__(256) void embed_gemm(const float* __restrict__ x,
                                                  const float* __restrict__ means,
                                                  const float* __restrict__ stds,
                                                  const _Float16* __restrict__ Wh,
                                                  const float* __restrict__ pe,
                                                  float* __restrict__ h) {
    int wid = threadIdx.x >> 5, lane = threadIdx.x & 31;
    int tile = blockIdx.x * 8 + wid;   // 32768 tiles
    int nt = tile & 7, mt = tile >> 3;
    int row = mt * 16 + (lane & 15);   // (bc, p) row
    int bc = row >> 6, p = row & 63;
    int b = bc >> 5, c = bc & 31;
    float mu = means[bc], inv = 1.f / stds[bc];
    int g0 = (lane >> 4) * 8;
    v16h a;
    for (int i = 0; i < 16; i++) {
        int k = 2 * g0 + i;
        float v = 0.f;
        if (k < PLEN) {
            int t = p * STRIDE + k;
            if (t > LLEN - 1) t = LLEN - 1;            // edge padding
            v = (x[((size_t)b * LLEN + t) * NCIN + c] - mu) * inv;
        }
        a[i] = (_Float16)v;
    }
    int cc = lane & 15, kb = (lane >> 4) * 16;
    v16h bm;
    for (int i = 0; i < 16; i++) {
        int k = kb + i;
        bm[i] = (k < PLEN) ? Wh[k * DMODEL + nt * 16 + cc] : (_Float16)0.f;
    }
    v8f acc = {};
    acc = __builtin_amdgcn_wmma_f32_16x16x32_f16(false, a, false, bm, (short)0, acc, false, false);
    int r0 = (lane >> 4) * 8;
    for (int i = 0; i < 8; i++) {
        int grow = mt * 16 + r0 + i;
        int gp = grow & 63;
        int col = nt * 16 + cc;
        h[(size_t)grow * DMODEL + col] = acc[i] + pe[gp * DMODEL + col];
    }
}

// ---------------- LayerNorm over D per (bc,p) row: one wave per row ----------------
__global__ __launch_bounds__(256) void ln_kernel(const float* __restrict__ h,
                                                 float* __restrict__ z,
                                                 const float* __restrict__ g,
                                                 const float* __restrict__ bta) {
    int wid = threadIdx.x >> 5, lane = threadIdx.x & 31;
    size_t rowi = (size_t)blockIdx.x * 8 + wid;        // < 65536
    const float* rp = h + rowi * DMODEL;
    float v[4];
    float s = 0.f;
    for (int j = 0; j < 4; j++) { v[j] = rp[lane * 4 + j]; s += v[j]; }
    for (int o = 16; o; o >>= 1) s += __shfl_xor(s, o, 32);
    float mu = s * (1.f / DMODEL);
    float q = 0.f;
    for (int j = 0; j < 4; j++) { float d = v[j] - mu; q += d * d; }
    for (int o = 16; o; o >>= 1) q += __shfl_xor(q, o, 32);
    float rstd = rsqrtf(q * (1.f / DMODEL) + 1e-5f);
    float* zp = z + rowi * DMODEL;
    for (int j = 0; j < 4; j++) {
        int d = lane * 4 + j;
        zp[d] = (v[j] - mu) * rstd * g[d] + bta[d];
    }
}

// ---------------- per-channel causal conv over P + D_skip + GELU ----------------
__global__ __launch_bounds__(256) void conv_kernel(const float* __restrict__ z,
                                                   const float* __restrict__ Ke,
                                                   const float* __restrict__ dskip,
                                                   float* __restrict__ y) {
    __shared__ float zs[PNUM * DMODEL];   // 32 KB  zs[p*128+d]
    __shared__ float ks[PNUM * DMODEL];   // 32 KB  ks[s*128+d]  (transposed kernel)
    int bc = blockIdx.x;
    const float* zp = z + (size_t)bc * PNUM * DMODEL;
    for (int idx = threadIdx.x; idx < PNUM * DMODEL; idx += 256) {
        zs[idx] = zp[idx];
        int s = idx >> 7, d = idx & 127;
        ks[idx] = Ke[d * PNUM + s];
    }
    __syncthreads();
    int d = threadIdx.x & 127;
    int phalf = threadIdx.x >> 7;          // interleave p for load balance
    float dsk = dskip[d];
    float* yp = y + (size_t)bc * PNUM * DMODEL;
    for (int i = 0; i < 32; i++) {
        int p = phalf + 2 * i;
        float acc = 0.f;
        for (int s = 0; s <= p; s++)
            acc += ks[s * DMODEL + d] * zs[(p - s) * DMODEL + d];
        float u = zs[p * DMODEL + d];
        float val = acc + u * dsk;
        val = 0.5f * val * (1.f + erff(val * 0.70710678118654752f));  // exact GELU
        yp[p * DMODEL + d] = val;
    }
}

// ---------------- W_out GEMM + GLU + residual: (65536x128)@(128x256) ----------------
// Block = 64-row x 128-col output panel. W (128x256 f16 = 64KB) staged into LDS by
// the Tensor Data Mover (tensor_load_to_lds, TENSORcnt); A panel (64x128) staged
// cooperatively as f16. Each wave: nt = wid, 4 M-subtiles x {value,gate} = 8 accs,
// 32 WMMAs entirely out of LDS. Fused bias + sigmoid-gate + residual epilogue.
__global__ __launch_bounds__(256) void glu_gemm(const float* __restrict__ y,
                                                const _Float16* __restrict__ W,
                                                const float* __restrict__ bo,
                                                float* __restrict__ h) {
    __shared__ _Float16 Wl[DMODEL * 256];   // 64 KB
    __shared__ _Float16 Al[64 * DMODEL];    // 16 KB
    int tid = threadIdx.x;
    int wid = tid >> 5, lane = tid & 31;
    size_t row0 = (size_t)blockIdx.x * 64;

    if (wid == 0) {
        // --- TDM descriptor (D#) for a 2D tile: 128 rows x 256 f16, contiguous ---
        unsigned int lds_off = (unsigned int)(size_t)(&Wl[0]);   // addr[31:0] = LDS offset
        unsigned long long ga = (unsigned long long)(size_t)W;
        u32x4 g0;
        g0.x = 1u;                                   // count=1, user descriptor
        g0.y = lds_off;                              // lds_addr
        g0.z = (unsigned int)ga;                     // global_addr[31:0]
        g0.w = (unsigned int)((ga >> 32) & 0x01FFFFFFu) | 0x80000000u;  // ga[56:32] | type=2
        i32x8 g1;
        g1[0] = 0x00010000;                          // workgroup_mask=0, data_size=1 (2B)
        g1[1] = (int)(256u << 16);                   // tensor_dim0[15:0]=256 at bits 63:48
        g1[2] = (int)(128u << 16);                   // tensor_dim1[15:0]=128 at bits 95:80
        g1[3] = (int)(256u << 16);                   // tile_dim0=256 at bits 127:112
        g1[4] = 128;                                 // tile_dim1=128 (tile_dim2=0)
        g1[5] = 256;                                 // tensor_dim0_stride[31:0]=256
        g1[6] = 0;
        g1[7] = 0;
        i32x4 gz = {0, 0, 0, 0};                     // groups 2/3 unused (<=2D tensor)
#if __has_include(<hip/amd_detail/amd_gfx1250_TDM.h>)
        i32x8 gz8 = {0, 0, 0, 0, 0, 0, 0, 0};
        __builtin_amdgcn_tensor_load_to_lds(g0, g1, gz, gz, gz8, 0);
#else
        __builtin_amdgcn_tensor_load_to_lds(g0, g1, gz, gz, 0);
#endif
    }
    // cooperative A stage: 64 rows x 128 f32 -> f16 (coalesced global reads)
    const float* ysrc = y + row0 * DMODEL;
    for (int idx = tid; idx < 64 * DMODEL; idx += 256)
        Al[idx] = (_Float16)ysrc[idx];
    __builtin_amdgcn_s_wait_tensorcnt(0);            // wave0's TDM; no-op for others
    __syncthreads();

    int nt = wid;                                    // column tile 0..7
    int cc = lane & 15, kb = (lane >> 4) * 16;
    int g0off = (lane >> 4) * 8;
    int ar = lane & 15;
    v8f acc1[4] = {}, acc2[4] = {};
    for (int kc = 0; kc < DMODEL; kc += 32) {
        v16h b1, b2;
        const _Float16* wp = &Wl[(kc + kb) * 256 + nt * 16 + cc];
        for (int i = 0; i < 16; i++) {
            b1[i] = wp[i * 256];
            b2[i] = wp[i * 256 + 128];
        }
        for (int sm = 0; sm < 4; sm++) {
            v16h a;
            const _Float16* ap = &Al[(sm * 16 + ar) * DMODEL + kc + kb];
            for (int i = 0; i < 8; i++) a[i] = ap[i];
            for (int i = 0; i < 8; i++) a[8 + i] = ap[8 + i];
            acc1[sm] = __builtin_amdgcn_wmma_f32_16x16x32_f16(false, a, false, b1, (short)0, acc1[sm], false, false);
            acc2[sm] = __builtin_amdgcn_wmma_f32_16x16x32_f16(false, a, false, b2, (short)0, acc2[sm], false, false);
        }
    }
    int r0 = (lane >> 4) * 8;
    int col = nt * 16 + cc;
    float b1v = bo[col], b2v = bo[128 + col];
    for (int sm = 0; sm < 4; sm++) {
        for (int i = 0; i < 8; i++) {
            size_t grow = row0 + sm * 16 + r0 + i;
            float v1 = acc1[sm][i] + b1v, v2 = acc2[sm][i] + b2v;
            float glu = v1 * (1.f / (1.f + expf(-v2)));
            h[grow * DMODEL + col] += glu;           // residual
        }
    }
}

// ---------------- head GEMM: (1024 x 8192)@(8192 x 96) + bias + RevIN denorm ----------------
__global__ __launch_bounds__(256) void head_gemm(const float* __restrict__ h,
                                                 const _Float16* __restrict__ W,
                                                 const float* __restrict__ bh,
                                                 const float* __restrict__ means,
                                                 const float* __restrict__ stds,
                                                 float* __restrict__ out) {
    int wid = threadIdx.x >> 5, lane = threadIdx.x & 31;
    int tile = blockIdx.x * 8 + wid;   // 384 tiles exactly (48 blocks x 8 waves)
    int nt = tile % 6, mt = tile / 6;
    int arow = mt * 16 + (lane & 15);  // bc row
    int g0 = (lane >> 4) * 8;
    int cc = lane & 15, kb = (lane >> 4) * 16;
    const float* ap = h + (size_t)arow * (DMODEL * PNUM);
    v8f acc = {};
    for (int kc = 0; kc < DMODEL * PNUM; kc += 32) {
        v16h a, bm;
        for (int i = 0; i < 16; i++) {
            int k = kc + 2 * g0 + i;
            int d = k >> 6, p = k & 63;
            a[i] = (_Float16)ap[p * DMODEL + d];
        }
        const _Float16* wp = W + (size_t)(kc + kb) * NPRED + nt * 16 + cc;
        for (int i = 0; i < 16; i++) bm[i] = wp[(size_t)i * NPRED];
        acc = __builtin_amdgcn_wmma_f32_16x16x32_f16(false, a, false, bm, (short)0, acc, false, false);
    }
    int r0 = (lane >> 4) * 8;
    int pred = nt * 16 + cc;
    float bias = bh[pred];
    for (int i = 0; i < 8; i++) {
        int bc = mt * 16 + r0 + i;
        int b = bc >> 5, c = bc & 31;
        out[((size_t)b * NPRED + pred) * NCIN + c] = (acc[i] + bias) * stds[bc] + means[bc];
    }
}

extern "C" void kernel_launch(void* const* d_in, const int* in_sizes, int n_in,
                              void* d_out, int out_size, void* d_ws, size_t ws_size,
                              hipStream_t stream) {
    const float* x_enc      = (const float*)d_in[0];
    const float* W_embed    = (const float*)d_in[4];
    const float* ln_g       = (const float*)d_in[5];
    const float* ln_b       = (const float*)d_in[6];
    const float* log_dt     = (const float*)d_in[7];
    const float* A_real_log = (const float*)d_in[8];
    const float* A_imag     = (const float*)d_in[9];
    const float* C_re       = (const float*)d_in[10];
    const float* C_im       = (const float*)d_in[11];
    const float* D_skip     = (const float*)d_in[12];
    const float* W_out      = (const float*)d_in[13];
    const float* b_out      = (const float*)d_in[14];
    const float* W_head     = (const float*)d_in[15];
    const float* b_head     = (const float*)d_in[16];

    char* ws = (char*)d_ws;
    size_t off = 0;
    auto alloc = [&](size_t bytes) -> void* {
        void* p = ws + off;
        off = (off + bytes + 255) & ~(size_t)255;
        return p;
    };
    float*    means    = (float*)alloc(BC * 4);
    float*    stds     = (float*)alloc(BC * 4);
    float*    Kbuf     = (float*)alloc((size_t)ELAYERS * DMODEL * PNUM * 4);
    float*    pe       = (float*)alloc((size_t)PNUM * DMODEL * 4);
    float*    hbuf     = (float*)alloc((size_t)MROWS * DMODEL * 4);
    float*    zbuf     = (float*)alloc((size_t)MROWS * DMODEL * 4);
    float*    ybuf     = (float*)alloc((size_t)MROWS * DMODEL * 4);
    _Float16* Wh_embed = (_Float16*)alloc((size_t)PLEN * DMODEL * 2);
    _Float16* Wh_out   = (_Float16*)alloc((size_t)ELAYERS * DMODEL * 2 * DMODEL * 2);
    _Float16* Wh_head  = (_Float16*)alloc((size_t)DMODEL * PNUM * NPRED * 2);

    stats_kernel<<<BC, 256, 0, stream>>>(x_enc, means, stds);
    cvt_kernel<<<(PLEN * DMODEL + 255) / 256, 256, 0, stream>>>(W_embed, Wh_embed, PLEN * DMODEL);
    cvt_kernel<<<(ELAYERS * DMODEL * 256 + 255) / 256, 256, 0, stream>>>(W_out, Wh_out, ELAYERS * DMODEL * 256);
    cvt_kernel<<<(DMODEL * PNUM * NPRED + 255) / 256, 256, 0, stream>>>(W_head, Wh_head, DMODEL * PNUM * NPRED);
    pe_kernel<<<(PNUM * DMODEL + 255) / 256, 256, 0, stream>>>(pe);
    s4_kernel_precompute<<<(ELAYERS * DMODEL * PNUM + 255) / 256, 256, 0, stream>>>(
        log_dt, A_real_log, A_imag, C_re, C_im, Kbuf);

    embed_gemm<<<MROWS / 16 / 8 * 8, 256, 0, stream>>>(x_enc, means, stds, Wh_embed, pe, hbuf);

    for (int e = 0; e < ELAYERS; e++) {
        ln_kernel<<<MROWS / 8, 256, 0, stream>>>(hbuf, zbuf, ln_g + e * DMODEL, ln_b + e * DMODEL);
        conv_kernel<<<BC, 256, 0, stream>>>(zbuf, Kbuf + (size_t)e * DMODEL * PNUM,
                                            D_skip + e * DMODEL, ybuf);
        glu_gemm<<<MROWS / 64, 256, 0, stream>>>(ybuf, Wh_out + (size_t)e * DMODEL * 256,
                                                 b_out + e * 256, hbuf);
    }

    head_gemm<<<48, 256, 0, stream>>>(hbuf, Wh_head, b_head, means, stds, (float*)d_out);
}
